// EdgeMPNNLayer_7799660609777
// MI455X (gfx1250) — compile-verified, matrix operand order checked by reading, not run on hardware
//
#include <hip/hip_runtime.h>
#include <hip/hip_bf16.h>

// ---------------------------------------------------------------------------
// EdgeMPNN layer, fused for gfx1250 (MI455X): bf16 WMMA GEMMs, LDS-swizzled
// operand staging, global_prefetch_b8 for next-tile gather, hw fp32 atomics
// for scatter-add, fused residual+LayerNorm.
// ---------------------------------------------------------------------------

typedef __attribute__((ext_vector_type(16))) __bf16 v16bf;
typedef __attribute__((ext_vector_type(8)))  float  v8f;

#define HDIM   128
#define EDIM   32
#define NNODES 50000
#define NEDGES 640000

// ISA 16-bit A/B fragment layout (CDNA5 ISA): lane L = g*16 + row holds
//   j in 0..7  -> K = kt*32 + g*8 + j
//   j in 8..15 -> K = kt*32 + g*8 + 16 + (j-8)
// Inverse map: given k_local = k % 32, find (g, j).
__device__ __forceinline__ int swz_idx(int kt, int row, int k_local) {
    int g = (k_local >> 3) & 1;
    int j = (k_local & 7) + ((k_local >> 4) << 3);
    return ((kt * 32 + g * 16 + row) << 4) + j;
}

// Convert a row-major f32 weight [K x 128] into bf16 fragment-swizzled LDS
// so each lane's 16 B-values for one WMMA are 32 contiguous bytes.
__device__ __forceinline__ void load_weights_swz(const float* __restrict__ W,
                                                 __bf16* dst, int KT,
                                                 int tid, int nthreads) {
    int total = KT * 8 * 512;               // KT ktiles x 8 ntiles x 32 lanes x 16
    for (int idx = tid; idx < total; idx += nthreads) {
        int j    = idx & 15;
        int lane = (idx >> 4) & 31;
        int blk  = idx >> 9;
        int nt   = blk & 7;
        int kt   = blk >> 3;
        int g    = lane >> 4;
        int nlow = lane & 15;
        int k_local = (j < 8) ? (g * 8 + j) : (g * 8 + 16 + (j - 8));
        int k = kt * 32 + k_local;
        int n = nt * 16 + nlow;
        dst[idx] = (__bf16)W[k * 128 + n];
    }
}

__device__ __forceinline__ v8f wmma_bf16(v16bf a, v16bf b, v8f c) {
    // D = A(16x32 bf16) * B(32x16 bf16) + C(16x16 f32)
    return __builtin_amdgcn_wmma_f32_16x16x32_bf16(
        /*neg_a=*/false, a, /*neg_b=*/false, b,
        /*c_mod=*/(short)0, c, /*reuse_a=*/false, /*reuse_b=*/false);
}

// Guard against compiler reordering around cross-lane LDS communication.
// DS ops are in-order within a wave; this pins the program order.
__device__ __forceinline__ void lds_fence() {
    asm volatile("s_wait_dscnt 0" ::: "memory");
}

// ---------------------------------------------------------------------------
__global__ void zero_ws_kernel(float4* __restrict__ p, long n4) {
    long i = (long)blockIdx.x * blockDim.x + threadIdx.x;
    long stride = (long)gridDim.x * blockDim.x;
    for (; i < n4; i += stride) p[i] = make_float4(0.f, 0.f, 0.f, 0.f);
}

// ---------------------------------------------------------------------------
// Edge pass: m = relu([h_src|h_dst|ea] @ ew1 + eb1) @ ew2 + eb2 ; atomics into agg
// One wave handles a 16-edge tile: 9x8 + 4x8 WMMA.
// ---------------------------------------------------------------------------
__global__ __launch_bounds__(256) void edge_msg_kernel(
    const float* __restrict__ h, const int* __restrict__ eidx,
    const float* __restrict__ eattr,
    const float* __restrict__ ew1, const float* __restrict__ eb1,
    const float* __restrict__ ew2, const float* __restrict__ eb2,
    float* __restrict__ agg)
{
    extern __shared__ __align__(16) char smem[];
    __bf16* w1s = (__bf16*)smem;            // 9*8*512  = 36864 bf16
    __bf16* w2s = w1s + 36864;              // 4*8*512  = 16384 bf16
    float*  b1s = (float*)(w2s + 16384);    // 128 f32
    float*  b2s = b1s + 128;                // 128 f32
    __bf16* wavebase = (__bf16*)(b2s + 128);

    const int tid  = threadIdx.x;
    const int wave = tid >> 5;
    const int lane = tid & 31;
    __bf16* Asz = wavebase + wave * (4608 + 2048);  // 16x288 swizzled
    __bf16* Msz = Asz + 4608;                       // 16x128 swizzled

    load_weights_swz(ew1, w1s, 9, tid, blockDim.x);
    load_weights_swz(ew2, w2s, 4, tid, blockDim.x);
    for (int i = tid; i < 128; i += blockDim.x) { b1s[i] = eb1[i]; b2s[i] = eb2[i]; }
    __syncthreads();

    const int* __restrict__ srcI = eidx;
    const int* __restrict__ dstI = eidx + NEDGES;
    const int cl = lane & 15, gl = lane >> 4;
    const int ntiles = NEDGES / 16;                         // 40000
    const int gw = blockIdx.x * (blockDim.x >> 5) + wave;
    const int nw = gridDim.x * (blockDim.x >> 5);

    for (int tile = gw; tile < ntiles; tile += nw) {
        const int ebase = tile * 16;

        // ---- stage A = [h[src] | h[dst] | ea] (16 x 288) coalesced -> swizzled bf16
        for (int row = 0; row < 16; ++row) {
            const int e = ebase + row;
            const int s = srcI[e];
            const int d = dstI[e];
            const float* hs = h + s * HDIM;
            const float* hd = h + d * HDIM;
            const float* ea = eattr + e * EDIM;
#pragma unroll
            for (int it = 0; it < 4; ++it) {
                int k = lane + it * 32;
                Asz[swz_idx(k >> 5, row, k & 31)] = (__bf16)hs[k];
            }
#pragma unroll
            for (int it = 0; it < 4; ++it) {
                int k = lane + 128 + it * 32;
                Asz[swz_idx(k >> 5, row, k & 31)] = (__bf16)hd[k - 128];
            }
            {
                int k = lane + 256;
                Asz[swz_idx(k >> 5, row, k & 31)] = (__bf16)ea[lane];
            }
        }
        lds_fence();

        // ---- prefetch next tile's gather targets while WMMAs run below
        // (global_prefetch_b8: no VGPR return, no counter; hides L2 gather latency)
        {
            int tnext = tile + nw;
            if (tnext < ntiles) {
                int e2   = tnext * 16 + cl;
                int idx2 = (gl == 0) ? srcI[e2] : dstI[e2];   // 32 lanes -> 16 src + 16 dst rows
                const char* hp = (const char*)(h + idx2 * HDIM);
#pragma unroll
                for (int ln = 0; ln < 4; ++ln)                // 512 B row = 4 cache lines
                    __builtin_prefetch(hp + ln * 128, 0, 3);
                // edge_attr slab for next tile: 16 * 128 B
                __builtin_prefetch((const char*)(eattr + (tnext * 16 + lane) * EDIM), 0, 3);
            }
        }

        // ---- GEMM1: (16x288) x (288x128), K = 9 chunks of 32
        v8f acc[8];
#pragma unroll
        for (int nt = 0; nt < 8; ++nt) acc[nt] = (v8f){};
#pragma unroll
        for (int kt = 0; kt < 9; ++kt) {
            v16bf a = *(const v16bf*)(Asz + ((kt * 32 + lane) << 4));
#pragma unroll
            for (int nt = 0; nt < 8; ++nt) {
                v16bf b = *(const v16bf*)(w1s + (((kt * 8 + nt) * 32 + lane) << 4));
                acc[nt] = wmma_bf16(a, b, acc[nt]);
            }
        }

        // ---- bias + ReLU -> mid tile (bf16, swizzled for GEMM2 A-operand)
#pragma unroll
        for (int nt = 0; nt < 8; ++nt) {
            int col = nt * 16 + cl;
            float bb = b1s[col];
#pragma unroll
            for (int r = 0; r < 8; ++r) {
                int m = r + gl * 8;
                float v = acc[nt][r] + bb;
                v = v > 0.f ? v : 0.f;
                Msz[swz_idx(col >> 5, m, col & 31)] = (__bf16)v;
            }
        }
        lds_fence();

        // ---- GEMM2: (16x128) x (128x128), K = 4 chunks
        v8f acc2[8];
#pragma unroll
        for (int nt = 0; nt < 8; ++nt) acc2[nt] = (v8f){};
#pragma unroll
        for (int kt = 0; kt < 4; ++kt) {
            v16bf a = *(const v16bf*)(Msz + ((kt * 32 + lane) << 4));
#pragma unroll
            for (int nt = 0; nt < 8; ++nt) {
                v16bf b = *(const v16bf*)(w2s + (((kt * 8 + nt) * 32 + lane) << 4));
                acc2[nt] = wmma_bf16(a, b, acc2[nt]);
            }
        }

        // ---- bias + scatter-add into agg (hw global_atomic_add_f32, L2-resident)
        int de[8];
#pragma unroll
        for (int r = 0; r < 8; ++r) de[r] = dstI[ebase + gl * 8 + r];
#pragma unroll
        for (int nt = 0; nt < 8; ++nt) {
            int col = nt * 16 + cl;
            float bb = b2s[col];
#pragma unroll
            for (int r = 0; r < 8; ++r) {
                unsafeAtomicAdd(agg + de[r] * HDIM + col, acc2[nt][r] + bb);
            }
        }
    }
}

// ---------------------------------------------------------------------------
// Node pass: u = relu([h|agg] @ nw1 + nb1) @ nw2 + nb2 ; out = LN(h + u)
// ---------------------------------------------------------------------------
__global__ __launch_bounds__(256) void node_upd_kernel(
    const float* __restrict__ h, const float* __restrict__ agg,
    const float* __restrict__ nw1, const float* __restrict__ nb1,
    const float* __restrict__ nw2, const float* __restrict__ nb2,
    const float* __restrict__ gamma, const float* __restrict__ beta,
    float* __restrict__ out)
{
    extern __shared__ __align__(16) char smem[];
    __bf16* w1s = (__bf16*)smem;            // 8*8*512 = 32768 bf16
    __bf16* w2s = w1s + 32768;              // 4*8*512 = 16384 bf16
    float*  b1s = (float*)(w2s + 16384);
    float*  b2s = b1s + 128;
    float*  gms = b2s + 128;
    float*  bts = gms + 128;
    char*   wavebase = (char*)(bts + 128);

    const int tid  = threadIdx.x;
    const int wave = tid >> 5;
    const int lane = tid & 31;
    char*   wb   = wavebase + wave * (8192 + 4096 + 128);
    __bf16* Asz  = (__bf16*)wb;             // 16x256 swizzled (8192 B)
    float*  xbuf = (float*)wb;              // aliased after GEMM1 consumes A
    __bf16* Msz  = (__bf16*)(wb + 8192);    // 16x128 swizzled (4096 B)
    float*  stat = (float*)(wb + 8192 + 4096); // 16 x {mu, rsigma}

    load_weights_swz(nw1, w1s, 8, tid, blockDim.x);
    load_weights_swz(nw2, w2s, 4, tid, blockDim.x);
    for (int i = tid; i < 128; i += blockDim.x) {
        b1s[i] = nb1[i]; b2s[i] = nb2[i]; gms[i] = gamma[i]; bts[i] = beta[i];
    }
    __syncthreads();

    const int cl = lane & 15, gl = lane >> 4;
    const int ntiles = NNODES / 16;                         // 3125
    const int gw = blockIdx.x * (blockDim.x >> 5) + wave;
    const int nwv = gridDim.x * (blockDim.x >> 5);

    for (int tile = gw; tile < ntiles; tile += nwv) {
        const int nbase = tile * 16;

        // ---- stage A = [h | agg] (16 x 256) -> swizzled bf16
        for (int row = 0; row < 16; ++row) {
            const int node = nbase + row;
            const float* hp = h   + node * HDIM;
            const float* ap = agg + node * HDIM;
#pragma unroll
            for (int it = 0; it < 4; ++it) {
                int k = lane + it * 32;
                Asz[swz_idx(k >> 5, row, k & 31)] = (__bf16)hp[k];
            }
#pragma unroll
            for (int it = 0; it < 4; ++it) {
                int k = lane + 128 + it * 32;
                Asz[swz_idx(k >> 5, row, k & 31)] = (__bf16)ap[k - 128];
            }
        }
        lds_fence();

        // ---- prefetch next tile's h/agg rows (streamed, 16 rows x 512 B each)
        {
            int tnext = tile + nwv;
            if (tnext < ntiles) {
                const float* hp = h   + (tnext * 16 + cl) * HDIM;
                const float* ap = agg + (tnext * 16 + cl) * HDIM;
#pragma unroll
                for (int it = 0; it < 2; ++it) {
                    int off = (gl * 2 + it) * 32;             // 2 of 4 lines per lane-group
                    __builtin_prefetch((const char*)(hp + off), 0, 3);
                    __builtin_prefetch((const char*)(ap + off), 0, 3);
                }
            }
        }

        // ---- GEMM1: (16x256) x (256x128)
        v8f acc[8];
#pragma unroll
        for (int nt = 0; nt < 8; ++nt) acc[nt] = (v8f){};
#pragma unroll
        for (int kt = 0; kt < 8; ++kt) {
            v16bf a = *(const v16bf*)(Asz + ((kt * 32 + lane) << 4));
#pragma unroll
            for (int nt = 0; nt < 8; ++nt) {
                v16bf b = *(const v16bf*)(w1s + (((kt * 8 + nt) * 32 + lane) << 4));
                acc[nt] = wmma_bf16(a, b, acc[nt]);
            }
        }

        // ---- bias + ReLU -> mid
#pragma unroll
        for (int nt = 0; nt < 8; ++nt) {
            int col = nt * 16 + cl;
            float bb = b1s[col];
#pragma unroll
            for (int r = 0; r < 8; ++r) {
                int m = r + gl * 8;
                float v = acc[nt][r] + bb;
                v = v > 0.f ? v : 0.f;
                Msz[swz_idx(col >> 5, m, col & 31)] = (__bf16)v;
            }
        }
        lds_fence();

        // ---- GEMM2: (16x128) x (128x128)
        v8f acc2[8];
#pragma unroll
        for (int nt = 0; nt < 8; ++nt) acc2[nt] = (v8f){};
#pragma unroll
        for (int kt = 0; kt < 4; ++kt) {
            v16bf a = *(const v16bf*)(Msz + ((kt * 32 + lane) << 4));
#pragma unroll
            for (int nt = 0; nt < 8; ++nt) {
                v16bf b = *(const v16bf*)(w2s + (((kt * 8 + nt) * 32 + lane) << 4));
                acc2[nt] = wmma_bf16(a, b, acc2[nt]);
            }
        }

        // ---- x = h + u  -> xbuf (f32, aliases Asz which is now dead)
#pragma unroll
        for (int nt = 0; nt < 8; ++nt) {
            int col = nt * 16 + cl;
            float bb = b2s[col];
#pragma unroll
            for (int r = 0; r < 8; ++r) {
                int m = r + gl * 8;
                int node = nbase + m;
                xbuf[m * HDIM + col] = acc2[nt][r] + bb + h[node * HDIM + col];
            }
        }
        lds_fence();

        // ---- LayerNorm row stats (lanes 0..15, one row each)
        if (lane < 16) {
            float s = 0.f, ss = 0.f;
            for (int c = 0; c < HDIM; ++c) {
                float v = xbuf[lane * HDIM + c];
                s += v; ss += v * v;
            }
            float mu  = s * (1.f / HDIM);
            float var = ss * (1.f / HDIM) - mu * mu;
            stat[lane * 2]     = mu;
            stat[lane * 2 + 1] = rsqrtf(var + 1e-5f);
        }
        lds_fence();

        // ---- normalize + affine, coalesced store
        for (int row = 0; row < 16; ++row) {
            float mu = stat[row * 2];
            float rs = stat[row * 2 + 1];
#pragma unroll
            for (int it = 0; it < 4; ++it) {
                int c = lane + it * 32;
                out[(nbase + row) * HDIM + c] =
                    (xbuf[row * HDIM + c] - mu) * rs * gms[c] + bts[c];
            }
        }
    }
}

// ---------------------------------------------------------------------------
extern "C" void kernel_launch(void* const* d_in, const int* in_sizes, int n_in,
                              void* d_out, int out_size, void* d_ws, size_t ws_size,
                              hipStream_t stream) {
    const float* h     = (const float*)d_in[0];
    const int*   eidx  = (const int*)  d_in[1];
    const float* eattr = (const float*)d_in[2];
    const float* ew1   = (const float*)d_in[3];
    const float* eb1   = (const float*)d_in[4];
    const float* ew2   = (const float*)d_in[5];
    const float* eb2   = (const float*)d_in[6];
    const float* nw1   = (const float*)d_in[7];
    const float* nb1   = (const float*)d_in[8];
    const float* nw2   = (const float*)d_in[9];
    const float* nb2   = (const float*)d_in[10];
    const float* gamma = (const float*)d_in[11];
    const float* beta  = (const float*)d_in[12];
    float* out = (float*)d_out;
    float* agg = (float*)d_ws;   // N*HDIM f32 scratch for scatter-add

    // zero the aggregation buffer every call (graph-replay safe)
    long n4 = (long)NNODES * HDIM / 4;
    zero_ws_kernel<<<2048, 256, 0, stream>>>((float4*)agg, n4);

    // edge message pass + scatter
    size_t lds_edge = (36864 + 16384) * sizeof(__bf16)   // weights
                    + 256 * sizeof(float)                // biases
                    + 8 * (4608 + 2048) * sizeof(__bf16);// per-wave staging
    edge_msg_kernel<<<1024, 256, lds_edge, stream>>>(
        h, eidx, eattr, ew1, eb1, ew2, eb2, agg);

    // node update + residual + LayerNorm
    size_t lds_node = (32768 + 16384) * sizeof(__bf16)
                    + 512 * sizeof(float)
                    + 8 * (8192 + 4096 + 128);
    node_upd_kernel<<<512, 256, lds_node, stream>>>(
        h, agg, nw1, nb1, nw2, nb2, gamma, beta, out);
}
